// LSTMCell_16458314678399
// MI455X (gfx1250) — matrix operand order repaired and checked by reading, not compiled, not run on hardware
//
#include <hip/hip_runtime.h>

#define LSTM_IN   2048
#define LSTM_OUT  2048
#define LSTM_B    4096

typedef __attribute__((ext_vector_type(16))) __bf16 v16bf;
typedef __attribute__((ext_vector_type(8)))  float  v8f;
typedef __attribute__((ext_vector_type(4)))  int    v4i;

#define AS1 __attribute__((address_space(1)))
#define AS3 __attribute__((address_space(3)))

// ---------------- async staging primitives (CDNA5 ASYNCcnt path) ----------------
#if __has_builtin(__builtin_amdgcn_global_load_async_to_lds_b128)
#define ASYNC_COPY_OK 1
#define COPY16(dst, src) \
    __builtin_amdgcn_global_load_async_to_lds_b128((AS1 v4i*)(src), (AS3 v4i*)(dst), 0, 0)
#else
#define ASYNC_COPY_OK 0
#define COPY16(dst, src) (*(uint4*)(dst) = *(const uint4*)(src))
#endif

#if ASYNC_COPY_OK
#if __has_builtin(__builtin_amdgcn_s_wait_asynccnt)
#define WAIT_ASYNC(n) __builtin_amdgcn_s_wait_asynccnt(n)
#else
#define WAIT_ASYNC(n) asm volatile("s_wait_asynccnt %0" ::"n"(n) : "memory")
#endif
#else
#define WAIT_ASYNC(n) do {} while (0)
#endif

// round-to-nearest-even f32 -> bf16, two values packed into one dword
__device__ __forceinline__ unsigned int pack_bf16(float lo, float hi) {
    unsigned int ul = __float_as_uint(lo);
    unsigned int uh = __float_as_uint(hi);
    ul = ul + 0x7FFFu + ((ul >> 16) & 1u);
    uh = uh + 0x7FFFu + ((uh >> 16) & 1u);
    return (ul >> 16) | (uh & 0xFFFF0000u);
}

__device__ __forceinline__ unsigned short f2bf(float f) {
    unsigned int u = __float_as_uint(f);
    u = u + 0x7FFFu + ((u >> 16) & 1u);
    return (unsigned short)(u >> 16);
}

// load a 32-byte WMMA fragment as two ds_load_b128 at +0 and +delta
__device__ __forceinline__ v16bf ld_frag(const unsigned char* p, int delta) {
    union { uint4 u[2]; v16bf v; } f;
    f.u[0] = *(const uint4*)(p);
    f.u[1] = *(const uint4*)(p + delta);
    return f.v;
}

__device__ __forceinline__ float sigmoidf_fast(float x) {
    return 1.0f / (1.0f + __expf(-x));
}

// ================= pre-pass 1: elementwise f32 -> bf16 (row-major kept) =========
__global__ __launch_bounds__(256)
void lstm_cvt_bf16_kernel(const float* __restrict__ in, unsigned short* __restrict__ out) {
    const size_t i = ((size_t)blockIdx.x * 256 + threadIdx.x) * 8;
    const float4 a = *(const float4*)(in + i);
    const float4 b = *(const float4*)(in + i + 4);
    uint4 r;
    r.x = pack_bf16(a.x, a.y);
    r.y = pack_bf16(a.z, a.w);
    r.z = pack_bf16(b.x, b.y);
    r.w = pack_bf16(b.z, b.w);
    *(uint4*)(out + i) = r;
}

// ====== pre-pass 2: W[k][n] f32 -> WT[n][k] bf16 (32x32 tiles through LDS) ======
__global__ __launch_bounds__(256)
void lstm_transpose_bf16_kernel(const float* __restrict__ W, unsigned short* __restrict__ WT) {
    __shared__ __align__(8) unsigned short tile[32][40];  // 80B row stride
    const int bn = blockIdx.x * 32;  // n (cols of W, rows of WT)
    const int bk = blockIdx.y * 32;  // k (rows of W, cols of WT)
    const int r  = threadIdx.x >> 3;  // 0..31
    const int c4 = threadIdx.x & 7;   // 0..7
    const float4 v = *(const float4*)(W + (size_t)(bk + r) * LSTM_OUT + bn + c4 * 4);
    const float vv[4] = { v.x, v.y, v.z, v.w };
    #pragma unroll
    for (int j = 0; j < 4; ++j) tile[c4 * 4 + j][r] = f2bf(vv[j]);
    __syncthreads();
    *(uint2*)(WT + (size_t)(bn + r) * LSTM_IN + bk + c4 * 4) = *(const uint2*)&tile[r][c4 * 4];
}

// ===================== main fused LSTM GEMM (bf16 WMMA) =========================
// LDS layout: A tiles (64x32 bf16, 80B row stride) double buffered at 0 / 5120,
// B tiles (4 gates x 64 cols x 32 K bf16, [col][k], 80B stride) at 10240 + buf*20480.
#define LDS_A(buf)        ((buf) * 5120)
#define LDS_B(buf, g)     (10240 + (buf) * 20480 + (g) * 5120)

__global__ __launch_bounds__(256)
void LSTMCell_16458314678399_kernel(
    const unsigned short* __restrict__ Xbf, const unsigned short* __restrict__ Hbf,
    const unsigned short* __restrict__ WxTf, const unsigned short* __restrict__ WxTg,
    const unsigned short* __restrict__ WxTi, const unsigned short* __restrict__ WxTo,
    const unsigned short* __restrict__ WhTf, const unsigned short* __restrict__ WhTg,
    const unsigned short* __restrict__ WhTi, const unsigned short* __restrict__ WhTo,
    const float* __restrict__ C,
    const float* __restrict__ b_f, const float* __restrict__ b_g,
    const float* __restrict__ b_i, const float* __restrict__ b_o,
    float* __restrict__ out)
{
    __shared__ __align__(16) unsigned char lds[2 * 5120 + 2 * 20480];

    const int tid    = threadIdx.x;
    const int lane   = tid & 31;
    const int wave   = tid >> 5;
    const int wave_m = wave >> 1;    // 0..3
    const int wave_n = wave & 1;     // 0..1
    const int m0     = blockIdx.y * 64;
    const int n0     = blockIdx.x * 64;
    const int mrow   = lane & 15;
    const int hlf    = lane >> 4;

    // staging addressing: each thread copies one 16B segment per tile
    const int srow = tid >> 2;   // 0..63 (A row / B col)
    const int sseg = tid & 3;    // 0..3

    v8f acc[4][2] = {};

    const int aBase = (wave_m * 16 + mrow) * 80 + hlf * 32;
    int bBase[2];
    bBase[0] = (wave_n * 32 + mrow) * 80 + hlf * 32;
    bBase[1] = bBase[0] + 16 * 80;

    for (int phase = 0; phase < 2; ++phase) {
        const unsigned short* Ap = phase ? Hbf : Xbf;
        const unsigned short* W[4];
        if (phase == 0) { W[0] = WxTf; W[1] = WxTg; W[2] = WxTi; W[3] = WxTo; }
        else            { W[0] = WhTf; W[1] = WhTg; W[2] = WhTi; W[3] = WhTo; }

        const int ldsOff = srow * 80 + sseg * 16;

        // running source pointers (advance by 32 bf16 per K-step)
        const unsigned short* aCur = Ap + (size_t)(m0 + srow) * LSTM_IN + sseg * 8;
        const unsigned short* bCur[4];
        #pragma unroll
        for (int g = 0; g < 4; ++g)
            bCur[g] = W[g] + (size_t)(n0 + srow) * LSTM_IN + sseg * 8;

        // prologue: stage tile k0=0 into buffer 0
        COPY16(lds + LDS_A(0) + ldsOff, aCur);
        #pragma unroll
        for (int g = 0; g < 4; ++g)
            COPY16(lds + LDS_B(0, g) + ldsOff, bCur[g]);
        aCur += 32;
        #pragma unroll
        for (int g = 0; g < 4; ++g) bCur[g] += 32;

        for (int k0 = 0; k0 < LSTM_IN; k0 += 32) {
            const int buf = (k0 >> 5) & 1;

            if (k0 + 32 < LSTM_IN) {
                // prefetch next tile into the other buffer (5 async b128 per thread)
                COPY16(lds + LDS_A(buf ^ 1) + ldsOff, aCur);
                #pragma unroll
                for (int g = 0; g < 4; ++g)
                    COPY16(lds + LDS_B(buf ^ 1, g) + ldsOff, bCur[g]);
                aCur += 32;
                #pragma unroll
                for (int g = 0; g < 4; ++g) bCur[g] += 32;
                WAIT_ASYNC(5);   // in-order: current tile's 5 copies have landed
            } else {
                WAIT_ASYNC(0);
            }
            __syncthreads();     // all waves' copies for this tile are visible

            // ---- batch all fragment loads, then issue 8 WMMAs back-to-back ----
            const v16bf afrag = ld_frag(lds + LDS_A(buf) + aBase, 16);
            v16bf bfr[8];
            #pragma unroll
            for (int g = 0; g < 4; ++g) {
                #pragma unroll
                for (int t = 0; t < 2; ++t)
                    bfr[g * 2 + t] = ld_frag(lds + LDS_B(buf, g) + bBase[t], 16);
            }
            #pragma unroll
            for (int g = 0; g < 4; ++g) {
                #pragma unroll
                for (int t = 0; t < 2; ++t)
                    acc[g][t] = __builtin_amdgcn_wmma_f32_16x16x32_bf16(
                        false, afrag, false, bfr[g * 2 + t], (short)0, acc[g][t],
                        false, false);
            }
            __syncthreads();     // done reading buf before it is overwritten
        }
    }

    // ---- fused epilogue: bias + activations + cell update ----
    const size_t OY = (size_t)LSTM_B * LSTM_OUT;
    #pragma unroll
    for (int t = 0; t < 2; ++t) {
        const int col = n0 + wave_n * 32 + t * 16 + mrow;
        const float vbf = b_f[col], vbg = b_g[col], vbi = b_i[col], vbo = b_o[col];
        #pragma unroll
        for (int i = 0; i < 8; ++i) {
            const int row = m0 + wave_m * 16 + hlf * 8 + i;
            const size_t idx = (size_t)row * LSTM_OUT + col;
            const float f  = sigmoidf_fast(acc[0][t][i] + vbf);
            const float g  = tanhf(acc[1][t][i] + vbg);
            const float ii = sigmoidf_fast(acc[2][t][i] + vbi);
            const float o  = sigmoidf_fast(acc[3][t][i] + vbo);
            const float cn = g * ii + f * C[idx];
            out[idx]      = cn;
            out[OY + idx] = o * tanhf(cn);
        }
    }
}

// ================= fallback (f32 inputs, in-loop conversion) ====================
__global__ __launch_bounds__(256)
void LSTMCell_16458314678399_kernel_fallback(
    const float* __restrict__ X,   const float* __restrict__ C,
    const float* __restrict__ H,
    const float* __restrict__ Wxf, const float* __restrict__ Wxg,
    const float* __restrict__ Wxi, const float* __restrict__ Wxo,
    const float* __restrict__ Whf, const float* __restrict__ Whg,
    const float* __restrict__ Whi, const float* __restrict__ Who,
    const float* __restrict__ b_f, const float* __restrict__ b_g,
    const float* __restrict__ b_i, const float* __restrict__ b_o,
    float* __restrict__ out)
{
    __shared__ __align__(16) unsigned char ldsA[64 * 80];
    __shared__ __align__(16) unsigned char ldsB[4 * 64 * 80];

    const int tid    = threadIdx.x;
    const int lane   = tid & 31;
    const int wave   = tid >> 5;
    const int wave_m = wave >> 1;
    const int wave_n = wave & 1;
    const int m0     = blockIdx.y * 64;
    const int n0     = blockIdx.x * 64;
    const int mrow   = lane & 15;
    const int hlf    = lane >> 4;

    v8f acc[4][2] = {};
    const int aBase = (wave_m * 16 + mrow) * 80 + hlf * 32;
    int bBase[2];
    bBase[0] = (wave_n * 32 + mrow) * 80 + hlf * 32;
    bBase[1] = bBase[0] + 16 * 80;

    for (int phase = 0; phase < 2; ++phase) {
        const float* Ap = phase ? H : X;
        const float* W[4];
        if (phase == 0) { W[0] = Wxf; W[1] = Wxg; W[2] = Wxi; W[3] = Wxo; }
        else            { W[0] = Whf; W[1] = Whg; W[2] = Whi; W[3] = Who; }

        for (int k0 = 0; k0 < LSTM_IN; k0 += 32) {
            __syncthreads();
            for (int q = tid; q < 512; q += 256) {
                const int row = q >> 3, c4 = q & 7;
                const float4 a = *(const float4*)(Ap + (size_t)(m0 + row) * LSTM_IN + k0 + c4 * 4);
                uint2 p;
                p.x = pack_bf16(a.x, a.y);
                p.y = pack_bf16(a.z, a.w);
                *(uint2*)(ldsA + row * 80 + c4 * 8) = p;
            }
            {
                const int k2 = tid >> 4;
                const int c4 = tid & 15;
                #pragma unroll
                for (int g = 0; g < 4; ++g) {
                    const float* wp = W[g] + (size_t)(k0 + 2 * k2) * LSTM_OUT + n0 + c4 * 4;
                    const float4 wa = *(const float4*)(wp);
                    const float4 wb = *(const float4*)(wp + LSTM_OUT);
                    const float wax[4] = { wa.x, wa.y, wa.z, wa.w };
                    const float wbx[4] = { wb.x, wb.y, wb.z, wb.w };
                    unsigned char* bgp = ldsB + g * 5120;
                    #pragma unroll
                    for (int j = 0; j < 4; ++j)
                        *(unsigned int*)(bgp + (c4 * 4 + j) * 80 + k2 * 4) =
                            pack_bf16(wax[j], wbx[j]);
                }
            }
            __syncthreads();

            const v16bf afrag = ld_frag(ldsA + aBase, 16);
            #pragma unroll
            for (int g = 0; g < 4; ++g) {
                #pragma unroll
                for (int t = 0; t < 2; ++t) {
                    const v16bf bfrag = ld_frag(ldsB + g * 5120 + bBase[t], 16);
                    acc[g][t] = __builtin_amdgcn_wmma_f32_16x16x32_bf16(
                        false, afrag, false, bfrag, (short)0, acc[g][t], false, false);
                }
            }
        }
    }

    const size_t OY = (size_t)LSTM_B * LSTM_OUT;
    #pragma unroll
    for (int t = 0; t < 2; ++t) {
        const int col = n0 + wave_n * 32 + t * 16 + mrow;
        const float vbf = b_f[col], vbg = b_g[col], vbi = b_i[col], vbo = b_o[col];
        #pragma unroll
        for (int i = 0; i < 8; ++i) {
            const int row = m0 + wave_m * 16 + hlf * 8 + i;
            const size_t idx = (size_t)row * LSTM_OUT + col;
            const float f  = sigmoidf_fast(acc[0][t][i] + vbf);
            const float g  = tanhf(acc[1][t][i] + vbg);
            const float ii = sigmoidf_fast(acc[2][t][i] + vbi);
            const float o  = sigmoidf_fast(acc[3][t][i] + vbo);
            const float cn = g * ii + f * C[idx];
            out[idx]      = cn;
            out[OY + idx] = o * tanhf(cn);
        }
    }
}

// =============================== launcher =======================================
extern "C" void kernel_launch(void* const* d_in, const int* in_sizes, int n_in,
                              void* d_out, int out_size, void* d_ws, size_t ws_size,
                              hipStream_t stream) {
    (void)in_sizes; (void)n_in; (void)out_size;
    // setup_inputs order: X, C, H, W_xf, W_xg, W_xi, W_xo, W_hf, W_hg, W_hi, W_ho,
    //                     b_f, b_g, b_i, b_o
    const float* X   = (const float*)d_in[0];
    const float* C   = (const float*)d_in[1];
    const float* H   = (const float*)d_in[2];
    const float* Wx[4] = { (const float*)d_in[3], (const float*)d_in[4],
                           (const float*)d_in[5], (const float*)d_in[6] };
    const float* Wh[4] = { (const float*)d_in[7], (const float*)d_in[8],
                           (const float*)d_in[9], (const float*)d_in[10] };
    const float* b_f = (const float*)d_in[11];
    const float* b_g = (const float*)d_in[12];
    const float* b_i = (const float*)d_in[13];
    const float* b_o = (const float*)d_in[14];
    float* out = (float*)d_out;

    // workspace layout (bf16): Xbf(16MB) | Hbf(16MB) | WxT[4](8MB ea) | WhT[4](8MB ea)
    const size_t XH_BYTES = (size_t)LSTM_B * LSTM_IN * 2;       // 16,777,216
    const size_t W_BYTES  = (size_t)LSTM_IN * LSTM_OUT * 2;     // 8,388,608
    const size_t NEED     = 2 * XH_BYTES + 8 * W_BYTES;         // 100,663,296

    if (ws_size >= NEED) {
        unsigned char* ws = (unsigned char*)d_ws;
        unsigned short* Xbf = (unsigned short*)(ws);
        unsigned short* Hbf = (unsigned short*)(ws + XH_BYTES);
        unsigned short* WxT[4], *WhT[4];
        for (int g = 0; g < 4; ++g) {
            WxT[g] = (unsigned short*)(ws + 2 * XH_BYTES + (size_t)g * W_BYTES);
            WhT[g] = (unsigned short*)(ws + 2 * XH_BYTES + (size_t)(4 + g) * W_BYTES);
        }

        // pre-pass: convert activations, convert+transpose weights
        const int cvtBlocks = (int)((size_t)LSTM_B * LSTM_IN / (256 * 8));  // 4096
        lstm_cvt_bf16_kernel<<<cvtBlocks, 256, 0, stream>>>(X, Xbf);
        lstm_cvt_bf16_kernel<<<cvtBlocks, 256, 0, stream>>>(H, Hbf);
        dim3 tgrid(LSTM_OUT / 32, LSTM_IN / 32);
        for (int g = 0; g < 4; ++g) {
            lstm_transpose_bf16_kernel<<<tgrid, 256, 0, stream>>>(Wx[g], WxT[g]);
            lstm_transpose_bf16_kernel<<<tgrid, 256, 0, stream>>>(Wh[g], WhT[g]);
        }

        dim3 grid(LSTM_OUT / 64, LSTM_B / 64);
        LSTMCell_16458314678399_kernel<<<grid, 256, 0, stream>>>(
            Xbf, Hbf,
            WxT[0], WxT[1], WxT[2], WxT[3],
            WhT[0], WhT[1], WhT[2], WhT[3],
            C, b_f, b_g, b_i, b_o, out);
    } else {
        dim3 grid(LSTM_OUT / 64, LSTM_B / 64);
        LSTMCell_16458314678399_kernel_fallback<<<grid, 256, 0, stream>>>(
            X, C, H, Wx[0], Wx[1], Wx[2], Wx[3], Wh[0], Wh[1], Wh[2], Wh[3],
            b_f, b_g, b_i, b_o, out);
    }
}
